// RNN_72816875537107
// MI455X (gfx1250) — matrix operand order repaired and checked by reading, not compiled, not run on hardware
//
#include <hip/hip_runtime.h>
#include <hip/hip_bf16.h>
#include <math.h>

// Problem dims (compile-time constants)
constexpr int BB = 64;    // batch
constexpr int TT = 1024;  // timesteps
constexpr int II = 64;    // input features
constexpr int HH = 512;   // hidden
constexpr int OO = 64;    // output features

typedef __attribute__((ext_vector_type(16))) __bf16 v16bf;
typedef __attribute__((ext_vector_type(8)))  __bf16 v8bf;
typedef __attribute__((ext_vector_type(8)))  float  v8f;

__device__ __forceinline__ v16bf pack16(v8bf lo, v8bf hi) {
  v16bf r;
#pragma unroll
  for (int e = 0; e < 8; ++e) { r[e] = lo[e]; r[e + 8] = hi[e]; }
  return r;
}

__device__ __forceinline__ v8f wmma_bf16(v16bf a, v16bf b, v8f c) {
  // D = A(16x32) * B(32x16) + C(16x16), f32 accumulate
  return __builtin_amdgcn_wmma_f32_16x16x32_bf16(false, a, false, b,
                                                 (short)0, c, false, false);
}

// ---------------------------------------------------------------------------
// Kernel 1: xp[b*T+t][h] = x[b,t,:] . W_ih[h,:] + b_ih[h] + b_hh[h]
// GEMM M=B*T=65536, N=H=512, K=I=64. One 16x16 tile per wave, K in 2 steps.
// ---------------------------------------------------------------------------
__global__ __launch_bounds__(256) void xp_gemm_kernel(
    const float* __restrict__ x, const float* __restrict__ W_ih,
    const float* __restrict__ b_ih, const float* __restrict__ b_hh,
    float* __restrict__ xp)
{
  const int wave = threadIdx.x >> 5, lane = threadIdx.x & 31;
  const int idx = blockIdx.x * 8 + wave;   // 131072 wave-tiles total
  const int tn  = idx & 31;                // H/16 = 32 n-tiles
  const int tm  = idx >> 5;                // (B*T)/16 = 4096 m-tiles
  const int rowA = tm * 16 + (lane & 15);
  const int akb  = (lane >> 4) * 16;       // A k sub-offset per half-wave
  const int nB   = tn * 16 + (lane & 15);
  const int bkb  = (lane >> 4) * 16;       // B k sub-offset per half-wave

  v8f acc = (v8f){0.f,0.f,0.f,0.f,0.f,0.f,0.f,0.f};
#pragma unroll
  for (int kk = 0; kk < 2; ++kk) {
    const float* ax = &x[(size_t)rowA * II + kk * 32 + akb];
    const float* bw = &W_ih[(size_t)nB * II + kk * 32 + bkb];
    v16bf a, b;
#pragma unroll
    for (int e = 0; e < 8; ++e) { a[e] = (__bf16)ax[e]; a[e + 8] = (__bf16)ax[e + 8]; }
#pragma unroll
    for (int e = 0; e < 16; ++e) b[e] = (__bf16)bw[e];
    acc = wmma_bf16(a, b, acc);
  }
  const int cn    = tn * 16 + (lane & 15);
  const float bias = b_ih[cn] + b_hh[cn];
  const int mbase = tm * 16 + (lane >> 4) * 8;
#pragma unroll
  for (int r = 0; r < 8; ++r)
    xp[(size_t)(mbase + r) * HH + cn] = acc[r] + bias;
}

// ---------------------------------------------------------------------------
// Kernel 2: serial recurrence h_t = tanh(xp_t + h_{t-1} W_hh^T), persistent
// single workgroup (32 waves on one WGP). h double-buffered in LDS (bf16),
// W_hh resident: k-steps 0..11 in VGPRs per wave, 12..15 in LDS.
// Also emits hs (bf16) for the output projection and h_last (f32) to d_out.
// ---------------------------------------------------------------------------
__global__ __launch_bounds__(1024) void rnn_recurrence_kernel(
    const float* __restrict__ xp, const float* __restrict__ h0,
    const float* __restrict__ W_hh, __bf16* __restrict__ hs,
    float* __restrict__ hlast)
{
  constexpr int KREG = 12;                      // k-steps with B-frags in VGPRs
  __shared__ __bf16 hbuf[2][BB][HH + 8];        // 133,120 B (padded: +4 banks/row)
  __shared__ __bf16 wlds[HH][128 + 8];          // 139,264 B: W_hh[:, 384:512]

  const int tid  = threadIdx.x;
  const int wave = tid >> 5, lane = tid & 31;

  // init h_0 into LDS buffer 0 (bf16)
  for (int i = tid; i < BB * HH; i += 1024)
    hbuf[0][i >> 9][i & (HH - 1)] = (__bf16)h0[i];
  // stage W_hh k=384..511 into LDS (bf16)
  for (int i = tid; i < HH * 128; i += 1024)
    wlds[i >> 7][i & 127] = (__bf16)W_hh[(size_t)(i >> 7) * HH + 384 + (i & 127)];

  // this wave owns output channels [wave*16, wave*16+16)
  const int nB  = wave * 16 + (lane & 15);
  const int bkb = (lane >> 4) * 16;
  v16bf bfrag[KREG];
#pragma unroll
  for (int kk = 0; kk < KREG; ++kk) {
    const float* src = &W_hh[(size_t)nB * HH + kk * 32 + bkb];
#pragma unroll
    for (int e = 0; e < 16; ++e) bfrag[kk][e] = (__bf16)src[e];
  }
  __syncthreads();

  const int arow = lane & 15;
  const int akb  = (lane >> 4) * 16;
  const int cm   = (lane >> 4) * 8;
  const int cn   = nB;

  for (int t = 0; t < TT; ++t) {
    const int cur = t & 1, nxt = cur ^ 1;

    // prefetch next step's xp slice so the epilogue load hits cache
    if (t + 1 < TT) {
#pragma unroll
      for (int m = 0; m < 4; ++m)
#pragma unroll
        for (int r = 0; r < 8; ++r)
          __builtin_prefetch(
              &xp[((size_t)(m * 16 + cm + r) * TT + (t + 1)) * HH + cn], 0, 0);
    }

    v8f acc[4];
#pragma unroll
    for (int m = 0; m < 4; ++m) acc[m] = (v8f){0.f,0.f,0.f,0.f,0.f,0.f,0.f,0.f};

#pragma unroll
    for (int kk = 0; kk < 16; ++kk) {
      v16bf b;
      if (kk < KREG) {
        b = bfrag[kk];
      } else {
        const int kw = kk * 32 + bkb - 384;
        b = pack16(*(const v8bf*)&wlds[nB][kw],
                   *(const v8bf*)&wlds[nB][kw + 8]);
      }
      const int k0 = kk * 32 + akb;
#pragma unroll
      for (int m = 0; m < 4; ++m) {
        const int row = m * 16 + arow;
        v16bf a = pack16(*(const v8bf*)&hbuf[cur][row][k0],
                         *(const v8bf*)&hbuf[cur][row][k0 + 8]);
        acc[m] = wmma_bf16(a, b, acc[m]);
      }
    }

#pragma unroll
    for (int m = 0; m < 4; ++m) {
#pragma unroll
      for (int r = 0; r < 8; ++r) {
        const int brow = m * 16 + cm + r;
        const float v  = acc[m][r] + xp[((size_t)brow * TT + t) * HH + cn];
        const float hv = tanhf(v);
        const __bf16 hb = (__bf16)hv;
        hbuf[nxt][brow][cn] = hb;
        hs[((size_t)brow * TT + t) * HH + cn] = hb;
        if (t == TT - 1) hlast[brow * HH + cn] = hv;
      }
    }
    __syncthreads();  // all reads of cur done; nxt fully written
  }
}

// ---------------------------------------------------------------------------
// Kernel 3: out[b*T+t][o] = tanh(hs[b*T+t,:] . W_out[o,:] + b_out[o])
// GEMM M=B*T=65536, N=O=64, K=H=512. One 16x16 tile per wave, 16 k-steps.
// ---------------------------------------------------------------------------
__global__ __launch_bounds__(256) void out_gemm_kernel(
    const __bf16* __restrict__ hs, const float* __restrict__ W_out,
    const float* __restrict__ b_out, float* __restrict__ out)
{
  const int wave = threadIdx.x >> 5, lane = threadIdx.x & 31;
  const int idx = blockIdx.x * 8 + wave;   // 16384 wave-tiles total
  const int tn  = idx & 3;                 // O/16 = 4
  const int tm  = idx >> 2;                // 4096
  const int rowA = tm * 16 + (lane & 15);
  const int akb  = (lane >> 4) * 16;
  const int nB   = tn * 16 + (lane & 15);
  const int bkb  = (lane >> 4) * 16;

  v8f acc = (v8f){0.f,0.f,0.f,0.f,0.f,0.f,0.f,0.f};
#pragma unroll
  for (int kk = 0; kk < 16; ++kk) {
    const int k0 = kk * 32;
    v16bf a = pack16(*(const v8bf*)&hs[(size_t)rowA * HH + k0 + akb],
                     *(const v8bf*)&hs[(size_t)rowA * HH + k0 + akb + 8]);
    const float* bw = &W_out[(size_t)nB * HH + k0 + bkb];
    v16bf b;
#pragma unroll
    for (int e = 0; e < 16; ++e) b[e] = (__bf16)bw[e];
    acc = wmma_bf16(a, b, acc);
  }
  const int cn    = tn * 16 + (lane & 15);
  const float bias = b_out[cn];
  const int mbase = tm * 16 + (lane >> 4) * 8;
#pragma unroll
  for (int r = 0; r < 8; ++r)
    out[(size_t)(mbase + r) * OO + cn] = tanhf(acc[r] + bias);
}

// ---------------------------------------------------------------------------
// Host launcher
// ---------------------------------------------------------------------------
extern "C" void kernel_launch(void* const* d_in, const int* in_sizes, int n_in,
                              void* d_out, int out_size, void* d_ws, size_t ws_size,
                              hipStream_t stream) {
  const float* x     = (const float*)d_in[0];   // [B,T,I]
  const float* h0    = (const float*)d_in[1];   // [1,B,H]
  const float* W_ih  = (const float*)d_in[2];   // [H,I]
  const float* W_hh  = (const float*)d_in[3];   // [H,H]
  const float* b_ih  = (const float*)d_in[4];   // [H]
  const float* b_hh  = (const float*)d_in[5];   // [H]
  const float* W_out = (const float*)d_in[6];   // [O,H]
  const float* b_out = (const float*)d_in[7];   // [O]

  float* out   = (float*)d_out;                       // [B,T,O]
  float* hlast = out + (size_t)BB * TT * OO;          // [1,B,H]

  // workspace: xp (f32, 128 MB) then hs (bf16, 64 MB)
  float*  xp = (float*)d_ws;
  __bf16* hs = (__bf16*)((char*)d_ws + (size_t)BB * TT * HH * sizeof(float));

  // 1) xp = x @ W_ih^T + b_ih + b_hh     (parallel GEMM)
  xp_gemm_kernel<<<((BB * TT / 16) * (HH / 16)) / 8, 256, 0, stream>>>(
      x, W_ih, b_ih, b_hh, xp);

  // 2) serial scan over T on one persistent WGP (weights register/LDS-resident)
  rnn_recurrence_kernel<<<1, 1024, 0, stream>>>(xp, h0, W_hh, hs, hlast);

  // 3) out = tanh(hs @ W_out^T + b_out)  (parallel GEMM)
  out_gemm_kernel<<<((BB * TT / 16) * (OO / 16)) / 8, 256, 0, stream>>>(
      hs, W_out, b_out, out);
}
